// ModifiedCPNN_20091857011217
// MI455X (gfx1250) — compile-verified, matrix operand order checked by reading, not run on hardware
//
#include <hip/hip_runtime.h>
#include <cstdint>
#include <cstddef>

// ---------------------------------------------------------------------------
// ModifiedCPNN forward for MI455X (gfx1250, wave32, WMMA, async global->LDS).
//   B=4096, D_IN=2048, H=4096, OUT=1000, T=0.5
// Compute-bound (~163 GFLOP vs ~150MB traffic @23.3TB/s): all GEMMs in bf16
// WMMA with fp32 accumulation; intermediates live in the 192MB L2.
// GEMM core: 64x128 block tile, 4 waves, wave tile 32x64, BK=64 (16 WMMA per
// wave per LDS buffer), double-buffered LDS fed by
// global_load_async_to_lds_b128 (ASYNCcnt), pipeline unrolled by 2 so buffer
// selects are literals and ds_load offsets are immediate; global addresses
// are mutating pointers (+128B per tile) to minimize address VALU and
// register pressure. N padded to tile multiples; ragged N=1000 handled only
// in the epilogue store guard.
// ---------------------------------------------------------------------------

#define B_DIM   4096
#define D_IN    2048
#define H_DIM   4096
#define OUT_DIM 1000
#define OUT_PAD 1024

typedef __attribute__((ext_vector_type(16))) __bf16 v16bf;
typedef __attribute__((ext_vector_type(8)))  float  v8f;

#define BM 64
#define BN 128
#define BK 64
#define GEMM_THREADS 128

#define A_TILE_ELEMS (BM * BK)          // 4096 ushort = 8KB
#define B_TILE_ELEMS (BN * BK)          // 8192 ushort = 16KB

__device__ __forceinline__ unsigned short f2bf(float f) {
  union { float f; unsigned int u; } c; c.f = f;
  unsigned int u = c.u;
  unsigned int r = u + 0x7fffu + ((u >> 16) & 1u);   // round-to-nearest-even
  return (unsigned short)(r >> 16);
}

union FragU { v16bf v; uint4 q[2]; };

// ISA 7.12.2 16-bit A-matrix 16x32 layout; B fragment uses identical indexing
// because weights are [N,K] row-major (lane c holds column c of B = row c of W).
// Loaded as 2x uint4 so the backend emits ds_load_b128.
template <int LDK>
__device__ __forceinline__ v16bf load_frag(const unsigned short* tile, int lane) {
  const int half = (lane >> 4) & 1;
  const int r    = lane & 15;
  const uint4* q = (const uint4*)(tile + r * LDK);
  FragU f;
  f.q[0] = q[half];        // K =  0..7  (+8*half)
  f.q[1] = q[2 + half];    // K = 16..23 (+8*half)
  return f.v;
}

// gfx1250 async global->LDS copy, 16B per lane, tracked by ASYNCcnt.
__device__ __forceinline__ void async_copy_b128(unsigned lds_addr,
                                                const unsigned short* gptr) {
  asm volatile("global_load_async_to_lds_b128 %0, %1, off"
               :: "v"(lds_addr), "v"((unsigned long long)(uintptr_t)gptr)
               : "memory");
}
__device__ __forceinline__ void wait_async_le12() {   // next tile still in flight
  asm volatile("s_wait_asynccnt 12" ::: "memory");
}
__device__ __forceinline__ void wait_async_0() {
  asm volatile("s_wait_asynccnt 0" ::: "memory");
}

// ---------------------------------------------------------------------------
// C[M,Nstore] = epilogue( A[M,K](bf16) @ W[Npad,K](bf16)^T )
//  DIST:  acc -> -2*sqrt(max(rowterm[m]+colterm[n]-2*acc, 1e-12))  (fp32)
//  else:  acc+bias[n], optional ReLU, store bf16 (Cb) or fp32 (Cf)
// Npad (= gridDim.y * BN) divides cleanly; Nstore may be ragged.
// K must be a multiple of 2*BK (holds: 2048, 4096, 2048, 1024).
// ---------------------------------------------------------------------------
template <bool DIST, bool RELU, bool STORE_BF16>
__global__ __launch_bounds__(GEMM_THREADS)
void wmma_gemm(const unsigned short* __restrict__ A,
               const unsigned short* __restrict__ Wt,
               const float* __restrict__ rowterm,   // x2[M]  (DIST only)
               const float* __restrict__ colterm,   // wsq[N] or bias[N]
               float* __restrict__ Cf,
               unsigned short* __restrict__ Cb,
               int M, int Nstore, int K) {
  __shared__ __attribute__((aligned(16)))
      unsigned short lds[2 * A_TILE_ELEMS + 2 * B_TILE_ELEMS];   // 48KB

  const int tid  = threadIdx.x;
  const int lane = tid & 31;
  const int wave = tid >> 5;          // 4 waves
  const int wr   = wave >> 1;         // 0..1 -> row offset wr*32
  const int wc   = wave & 1;          // 0..1 -> col offset wc*64
  const int bm   = blockIdx.x * BM;
  const int bn   = blockIdx.y * BN;

  unsigned short* As0 = &lds[0];
  unsigned short* Bs0 = &lds[2 * A_TILE_ELEMS];
  const unsigned ldsA = (unsigned)(uintptr_t)As0;   // LDS byte address
  const unsigned ldsB = (unsigned)(uintptr_t)Bs0;

  // per-thread async chunk assignment (16B chunks; 8 chunks per 64-elem row).
  // Pointers mutate (+BK per issued tile) -> steady loop has only +128B adds.
  const unsigned short* gA[4]; unsigned laOff[4];
#pragma unroll
  for (int j = 0; j < 4; ++j) {
    const int chunk = tid * 4 + j;           // 0..511
    const int row   = chunk >> 3;            // 0..63
    const int col   = (chunk & 7) * 8;       // 0..56
    gA[j]    = A + (size_t)(bm + row) * K + col;
    laOff[j] = (unsigned)((row * BK + col) * 2);
  }
  const unsigned short* gB[8]; unsigned lbOff[8];
#pragma unroll
  for (int j = 0; j < 8; ++j) {
    const int chunk = tid * 8 + j;           // 0..1023
    const int row   = chunk >> 3;            // 0..127
    const int col   = (chunk & 7) * 8;
    gB[j]    = Wt + (size_t)(bn + row) * K + col;
    lbOff[j] = (unsigned)((row * BK + col) * 2);
  }

  auto issue_tile = [&](int buf) {           // 12 async loads / thread
    const unsigned ab = ldsA + (unsigned)buf * (A_TILE_ELEMS * 2);
    const unsigned bb = ldsB + (unsigned)buf * (B_TILE_ELEMS * 2);
#pragma unroll
    for (int j = 0; j < 4; ++j) { async_copy_b128(ab + laOff[j], gA[j]); gA[j] += BK; }
#pragma unroll
    for (int j = 0; j < 8; ++j) { async_copy_b128(bb + lbOff[j], gB[j]); gB[j] += BK; }
  };

  v8f c[2][4];
#pragma unroll
  for (int i = 0; i < 2; ++i)
#pragma unroll
    for (int j = 0; j < 4; ++j)
      c[i][j] = (v8f){0.f, 0.f, 0.f, 0.f, 0.f, 0.f, 0.f, 0.f};

  auto compute_tile = [&](int buf) {         // 16 WMMA / wave
    const unsigned short* Ac = As0 + buf * A_TILE_ELEMS;
    const unsigned short* Bc = Bs0 + buf * B_TILE_ELEMS;
#pragma unroll
    for (int kh = 0; kh < 2; ++kh) {         // two 32-wide K halves
      v16bf a[2], b[4];
#pragma unroll
      for (int i = 0; i < 2; ++i)
        a[i] = load_frag<BK>(Ac + (wr * 32 + i * 16) * BK + kh * 32, lane);
#pragma unroll
      for (int j = 0; j < 4; ++j)
        b[j] = load_frag<BK>(Bc + (wc * 64 + j * 16) * BK + kh * 32, lane);
#pragma unroll
      for (int i = 0; i < 2; ++i)
#pragma unroll
        for (int j = 0; j < 4; ++j)
          c[i][j] = __builtin_amdgcn_wmma_f32_16x16x32_bf16(
              false, a[i], false, b[j], (short)0, c[i][j], false, false);
    }
  };

  const int ksteps = K / BK;                 // even, >= 16 for all our GEMMs

  // software pipeline, two tiles in flight, unrolled x2 (literal buf selects)
  issue_tile(0);                             // tile 0 -> buf0
  issue_tile(1);                             // tile 1 -> buf1
  for (int ks = 0; ks < ksteps - 2; ks += 2) {
    wait_async_le12();                       // tile ks resident
    __syncthreads();
    compute_tile(0);
    __syncthreads();                         // reads done before reuse of buf0
    issue_tile(0);                           // tile ks+2 -> buf0
    wait_async_le12();                       // tile ks+1 resident
    __syncthreads();
    compute_tile(1);
    __syncthreads();
    issue_tile(1);                           // tile ks+3 -> buf1
  }
  wait_async_le12();                         // tile ksteps-2
  __syncthreads();
  compute_tile(0);
  __syncthreads();
  wait_async_0();                            // tile ksteps-1
  __syncthreads();
  compute_tile(1);

  // epilogue: C/D layout -> VGPR v holds M = v + 8*(lane>=16), N = lane&15
  const int half = lane >> 4;
  const int nlo  = lane & 15;
#pragma unroll
  for (int i = 0; i < 2; ++i) {
#pragma unroll
    for (int j = 0; j < 4; ++j) {
#pragma unroll
      for (int v = 0; v < 8; ++v) {
        const int m = bm + wr * 32 + i * 16 + v + 8 * half;
        const int n = bn + wc * 64 + j * 16 + nlo;
        if (n >= Nstore) continue;           // only ragged for the final GEMM
        float acc = c[i][j][v];
        if (DIST) {
          float d2   = rowterm[m] + colterm[n] - 2.0f * acc;
          float dist = sqrtf(fmaxf(d2, 1e-12f));
          Cf[(size_t)m * Nstore + n] = -2.0f * dist;     // -dist / 0.5
        } else {
          float val = acc + colterm[n];
          if (RELU) val = fmaxf(val, 0.0f);
          if (STORE_BF16) Cb[(size_t)m * Nstore + n] = f2bf(val);
          else            Cf[(size_t)m * Nstore + n] = val;
        }
      }
    }
  }
  (void)M;
}

// ---------------------------------------------------------------------------
// fp32 -> bf16 with zero padding (pads w3 rows 1000..1023 so the GEMM loop
// needs no bounds checks).
__global__ void f32_to_bf16_pad_kernel(const float* __restrict__ in,
                                       unsigned short* __restrict__ out,
                                       size_t n_src, size_t n_dst) {
  size_t i = (size_t)blockIdx.x * blockDim.x + threadIdx.x;
  const size_t stride = (size_t)gridDim.x * blockDim.x;
  for (; i < n_dst; i += stride) out[i] = (i < n_src) ? f2bf(in[i]) : 0;
}

__global__ __launch_bounds__(256)
void rowsumsq_kernel(const float* __restrict__ X, float* __restrict__ out, int D) {
  const int row = blockIdx.x;
  const float* p = X + (size_t)row * D;
  float s = 0.f;
  for (int i = threadIdx.x; i < D; i += 256) { float v = p[i]; s += v * v; }
  __shared__ float sh[256];
  sh[threadIdx.x] = s;
  __syncthreads();
  for (int st = 128; st > 0; st >>= 1) {
    if (threadIdx.x < st) sh[threadIdx.x] += sh[threadIdx.x + st];
    __syncthreads();
  }
  if (threadIdx.x == 0) out[row] = sh[0];
}

// softmax over logits row + winner = argmax(logits) (== argmin distance,
// first-index tie-break); probs written as bf16 for the next WMMA GEMM.
__global__ __launch_bounds__(256)
void softmax_winner_kernel(const float* __restrict__ logits,
                           unsigned short* __restrict__ probs,
                           int* __restrict__ winners, int Hn) {
  const int row = blockIdx.x;
  const float* L = logits + (size_t)row * Hn;
  __shared__ float smax[256];
  __shared__ int   sidx[256];
  __shared__ float ssum[256];

  float best = -__builtin_huge_valf();
  int   bidx = 0x7fffffff;
  for (int i = threadIdx.x; i < Hn; i += 256) {
    float v = L[i];
    if (v > best || (v == best && i < bidx)) { best = v; bidx = i; }
  }
  smax[threadIdx.x] = best; sidx[threadIdx.x] = bidx;
  __syncthreads();
  for (int s = 128; s > 0; s >>= 1) {
    if (threadIdx.x < s) {
      float o = smax[threadIdx.x + s]; int oi = sidx[threadIdx.x + s];
      if (o > smax[threadIdx.x] ||
          (o == smax[threadIdx.x] && oi < sidx[threadIdx.x])) {
        smax[threadIdx.x] = o; sidx[threadIdx.x] = oi;
      }
    }
    __syncthreads();
  }
  const float rmax = smax[0];
  if (threadIdx.x == 0) winners[row] = sidx[0];

  float lsum = 0.f;
  for (int i = threadIdx.x; i < Hn; i += 256) lsum += __expf(L[i] - rmax);
  ssum[threadIdx.x] = lsum;
  __syncthreads();
  for (int s = 128; s > 0; s >>= 1) {
    if (threadIdx.x < s) ssum[threadIdx.x] += ssum[threadIdx.x + s];
    __syncthreads();
  }
  const float inv = 1.0f / ssum[0];
  unsigned short* P = probs + (size_t)row * Hn;
  for (int i = threadIdx.x; i < Hn; i += 256)
    P[i] = f2bf(__expf(L[i] - rmax) * inv);
}

// ---------------------------------------------------------------------------
static inline size_t align256(size_t x) { return (x + 255) & ~(size_t)255; }

extern "C" void kernel_launch(void* const* d_in, const int* in_sizes, int n_in,
                              void* d_out, int out_size, void* d_ws, size_t ws_size,
                              hipStream_t stream) {
  (void)in_sizes; (void)n_in; (void)out_size; (void)ws_size;

  const float* x   = (const float*)d_in[0];   // [4096,2048]
  const float* kw  = (const float*)d_in[1];   // [4096,2048]
  const float* w1  = (const float*)d_in[2];   // [2048,4096]
  const float* b1  = (const float*)d_in[3];   // [2048]
  const float* w2  = (const float*)d_in[4];   // [1024,2048]
  const float* b2  = (const float*)d_in[5];   // [1024]
  const float* w3  = (const float*)d_in[6];   // [1000,1024]
  const float* b3  = (const float*)d_in[7];   // [1000]

  float* out_f   = (float*)d_out;                                      // [4096,1000]
  int*   winners = (int*)((char*)d_out + (size_t)B_DIM * OUT_DIM * 4); // [4096]

  // ---- workspace layout
  char* ws = (char*)d_ws;
  size_t off = 0;
  auto carve = [&](size_t bytes) { char* p = ws + off; off = align256(off + bytes); return p; };
  unsigned short* xb  = (unsigned short*)carve((size_t)B_DIM * D_IN * 2);
  unsigned short* kwb = (unsigned short*)carve((size_t)H_DIM * D_IN * 2);
  unsigned short* w1b = (unsigned short*)carve((size_t)2048 * 4096 * 2);
  unsigned short* w2b = (unsigned short*)carve((size_t)1024 * 2048 * 2);
  unsigned short* w3b = (unsigned short*)carve((size_t)OUT_PAD * 1024 * 2);  // padded
  float*          x2  = (float*)carve((size_t)B_DIM * 4);
  float*          wsq = (float*)carve((size_t)H_DIM * 4);
  float*          lgt = (float*)carve((size_t)B_DIM * H_DIM * 4);
  unsigned short* prb = (unsigned short*)carve((size_t)B_DIM * H_DIM * 2);
  unsigned short* h1  = (unsigned short*)carve((size_t)B_DIM * 2048 * 2);
  unsigned short* h2  = (unsigned short*)carve((size_t)B_DIM * 1024 * 2);

  // ---- 1) fp32 -> bf16 conversions (w3 zero-padded to 1024 rows)
  f32_to_bf16_pad_kernel<<<2048, 256, 0, stream>>>(x,  xb,  (size_t)B_DIM * D_IN, (size_t)B_DIM * D_IN);
  f32_to_bf16_pad_kernel<<<2048, 256, 0, stream>>>(kw, kwb, (size_t)H_DIM * D_IN, (size_t)H_DIM * D_IN);
  f32_to_bf16_pad_kernel<<<2048, 256, 0, stream>>>(w1, w1b, (size_t)2048 * 4096, (size_t)2048 * 4096);
  f32_to_bf16_pad_kernel<<<1024, 256, 0, stream>>>(w2, w2b, (size_t)1024 * 2048, (size_t)1024 * 2048);
  f32_to_bf16_pad_kernel<<<1024, 256, 0, stream>>>(w3, w3b, (size_t)OUT_DIM * 1024, (size_t)OUT_PAD * 1024);

  // ---- 2) row sum-of-squares
  rowsumsq_kernel<<<B_DIM, 256, 0, stream>>>(x,  x2,  D_IN);
  rowsumsq_kernel<<<H_DIM, 256, 0, stream>>>(kw, wsq, D_IN);

  // ---- 3) GEMM1 + distance/logit epilogue: [4096,4096], K=2048
  {
    dim3 grid(B_DIM / BM, H_DIM / BN);
    wmma_gemm<true, false, false><<<grid, GEMM_THREADS, 0, stream>>>(
        xb, kwb, x2, wsq, lgt, nullptr, B_DIM, H_DIM, D_IN);
  }

  // ---- 4) softmax + winners
  softmax_winner_kernel<<<B_DIM, 256, 0, stream>>>(lgt, prb, winners, H_DIM);

  // ---- 5) h1 = relu(probs @ w1^T + b1) : [4096,2048], K=4096
  {
    dim3 grid(B_DIM / BM, 2048 / BN);
    wmma_gemm<false, true, true><<<grid, GEMM_THREADS, 0, stream>>>(
        prb, w1b, nullptr, b1, nullptr, h1, B_DIM, 2048, 4096);
  }
  // ---- 6) h2 = relu(h1 @ w2^T + b2) : [4096,1024], K=2048
  {
    dim3 grid(B_DIM / BM, 1024 / BN);
    wmma_gemm<false, true, true><<<grid, GEMM_THREADS, 0, stream>>>(
        h1, w2b, nullptr, b2, nullptr, h2, B_DIM, 1024, 2048);
  }
  // ---- 7) out = h2 @ w3^T + b3 : [4096,1000] fp32 (W padded to 1024 rows)
  {
    dim3 grid(B_DIM / BM, OUT_PAD / BN);
    wmma_gemm<false, false, false><<<grid, GEMM_THREADS, 0, stream>>>(
        h2, w3b, nullptr, b3, out_f, nullptr, B_DIM, OUT_DIM, 1024);
  }
}
